// GNN_node_Virtualnode_14482629722243
// MI455X (gfx1250) — matrix operand order repaired and checked by reading, not compiled, not run on hardware
//
#include <hip/hip_runtime.h>

// ---- problem constants (from reference) ----
#define NN   50000      // nodes
#define EE   600000     // edges
#define DD   256        // hidden dim
#define LL   5          // layers
#define GG   256        // graphs
#define NAF  9
#define AV   128
#define NBF  3
#define BV   8
#define BN_EPS 1e-5f

typedef __attribute__((ext_vector_type(16))) _Float16 v16h;
typedef __attribute__((ext_vector_type(8)))  _Float16 v8h;
typedef __attribute__((ext_vector_type(8)))  float    v8f;

// ------------------------- elementwise / gather kernels -------------------------

__global__ void k_zero(float* p, long n) {
    long i = (long)blockIdx.x * blockDim.x + threadIdx.x;
    if (i < n) p[i] = 0.f;
}

// h[n][c] = sum_f atom_emb[f][x[n][f]][c]
__global__ void k_atom_encode(const int* __restrict__ x,
                              const float* __restrict__ atom_emb,
                              float* __restrict__ h) {
    int n = blockIdx.x, c = threadIdx.x;
    float s = 0.f;
#pragma unroll
    for (int f = 0; f < NAF; ++f) {
        int id = x[n * NAF + f];
        s += atom_emb[((size_t)(f * AV + id)) * DD + c];
    }
    h[(size_t)n * DD + c] = s;
}

__global__ void k_vn_init(float* __restrict__ vn, const float* __restrict__ vn_emb) {
    vn[blockIdx.x * DD + threadIdx.x] = vn_emb[threadIdx.x];
}

__global__ void k_add_vn(float* __restrict__ h, const float* __restrict__ vn,
                         const int* __restrict__ batch) {
    int n = blockIdx.x, c = threadIdx.x;
    h[(size_t)n * DD + c] += vn[batch[n] * DD + c];
}

// msg = relu(h[src] + bond_emb_sum); atomicAdd into agg[dst]
__global__ void k_edge_msg(const int* __restrict__ ei, const int* __restrict__ ea,
                           const float* __restrict__ be,   // bond_emb layer base [3][8][256]
                           const float* __restrict__ h, float* __restrict__ agg) {
    int e = blockIdx.x, c = threadIdx.x;
    int s = ei[e];
    int d = ei[EE + e];
    float v = h[(size_t)s * DD + c];
#pragma unroll
    for (int f = 0; f < NBF; ++f) {
        int id = ea[e * NBF + f];
        v += be[(f * BV + id) * DD + c];
    }
    v = fmaxf(v, 0.f);
    atomicAdd(&agg[(size_t)d * DD + c], v);
}

// z = (1+eps[l])*h + agg, emitted directly as fp16 GEMM operand
__global__ void k_make_z_half(const float* __restrict__ h, const float* __restrict__ agg,
                              const float* __restrict__ eps, int l,
                              _Float16* __restrict__ a16) {
    size_t i = (size_t)blockIdx.x * blockDim.x + threadIdx.x;
    if (i >= (size_t)NN * DD) return;
    float z = (1.f + eps[l]) * h[i] + agg[i];
    a16[i] = (_Float16)z;
}

// WT[nc][k] = (half)W[k][nc]
__global__ void k_weightT_half(const float* __restrict__ W, _Float16* __restrict__ WT,
                               int K, int Nc) {
    long i = (long)blockIdx.x * blockDim.x + threadIdx.x;
    if (i >= (long)K * Nc) return;
    int nc = (int)(i / K), k = (int)(i % K);
    WT[i] = (_Float16)W[(size_t)k * Nc + nc];
}

// ------------------------- WMMA GEMM: C[M,Nc] = A16[M,K] @ WT16[Nc,K]^T + bias ----------------
// Register-tiled: each wave computes a 32x32 output block (2 M-tiles x 2 N-tiles,
// 4 f32 accumulators). Per K-step: 4 x B128-pair loads feed 4 WMMAs (32 B/WMMA).
// grid = (ceil(Mtiles/2), Nc/128), block = 128 (4 waves; wave w owns N pair blockIdx.y*8 + 2w)
__device__ __forceinline__ v16h ld_op(const _Float16* p, int stride2) {
    v8h lo = *(const v8h*)(p);
    v8h hi = *(const v8h*)(p + stride2);
    return __builtin_shufflevector(lo, hi, 0,1,2,3,4,5,6,7,8,9,10,11,12,13,14,15);
}

__global__ __launch_bounds__(128)
void k_wmma_gemm(const _Float16* __restrict__ A, const _Float16* __restrict__ BT,
                 const float* __restrict__ bias, float* __restrict__ C,
                 int M, int K, int Nc) {
    int wave = threadIdx.x >> 5;
    int lane = threadIdx.x & 31;
    int hi   = lane >> 4;            // 0: lanes 0-15, 1: lanes 16-31
    int lr   = lane & 15;

    int mt0 = blockIdx.x * 2;
    int mt1 = mt0 + 1;
    bool m1v = (mt1 * 16 < M);
    int mt1c = m1v ? mt1 : mt0;      // clamp (uniform): redundant compute, stores skipped
    int nt0 = blockIdx.y * 8 + wave * 2;
    int nt1 = nt0 + 1;

    // A 16-bit layout: lane half selects K chunks [+0..7] and [+16..23]
    const _Float16* Ar0 = A + (size_t)(mt0  * 16 + lr) * K + hi * 8;
    const _Float16* Ar1 = A + (size_t)(mt1c * 16 + lr) * K + hi * 8;
    // B layout: same as A, lane half selects K chunks [+0..7] and [+16..23]
    const _Float16* Br0 = BT + (size_t)(nt0 * 16 + lr) * K + hi * 8;
    const _Float16* Br1 = BT + (size_t)(nt1 * 16 + lr) * K + hi * 8;

    v8f acc00 = {}, acc01 = {}, acc10 = {}, acc11 = {};
    for (int k0 = 0; k0 < K; k0 += 32) {
        v16h a0 = ld_op(Ar0 + k0, 16);
        v16h a1 = ld_op(Ar1 + k0, 16);
        v16h b0 = ld_op(Br0 + k0, 16);
        v16h b1 = ld_op(Br1 + k0, 16);
        acc00 = __builtin_amdgcn_wmma_f32_16x16x32_f16(false, a0, false, b0, (short)0, acc00, false, false);
        acc01 = __builtin_amdgcn_wmma_f32_16x16x32_f16(false, a0, false, b1, (short)0, acc01, false, false);
        acc10 = __builtin_amdgcn_wmma_f32_16x16x32_f16(false, a1, false, b0, (short)0, acc10, false, false);
        acc11 = __builtin_amdgcn_wmma_f32_16x16x32_f16(false, a1, false, b1, (short)0, acc11, false, false);
    }

    int n0 = nt0 * 16 + lr;
    int n1 = nt1 * 16 + lr;
    float bv0 = bias ? bias[n0] : 0.f;
    float bv1 = bias ? bias[n1] : 0.f;
#pragma unroll
    for (int j = 0; j < 8; ++j) {                 // D layout: vgpr j -> row j + hi*8
        int r0 = mt0 * 16 + hi * 8 + j;
        C[(size_t)r0 * Nc + n0] = acc00[j] + bv0;
        C[(size_t)r0 * Nc + n1] = acc01[j] + bv1;
    }
    if (m1v) {
#pragma unroll
        for (int j = 0; j < 8; ++j) {
            int r1 = mt1 * 16 + hi * 8 + j;
            C[(size_t)r1 * Nc + n0] = acc10[j] + bv0;
            C[(size_t)r1 * Nc + n1] = acc11[j] + bv1;
        }
    }
}

// ------------------------- BatchNorm (training-mode, batch statistics) -------------------------

// per-block register accumulation over 64 rows, then one atomicAdd per column
__global__ void k_colstats(const float* __restrict__ x, int M, int C, float* __restrict__ sums) {
    int row0 = blockIdx.x * 64;
    int rows = (M - row0 < 64) ? (M - row0) : 64;
    for (int c = threadIdx.x; c < C; c += blockDim.x) {
        float s = 0.f, q = 0.f;
        for (int r = 0; r < rows; ++r) {
            float v = x[(size_t)(row0 + r) * C + c];
            s += v; q += v * v;
        }
        atomicAdd(&sums[c], s);
        atomicAdd(&sums[C + c], q);
    }
}

// scsh[c] = g*rstd ; scsh[C+c] = b - mean*g*rstd
__global__ void k_bn_finalize(const float* __restrict__ sums, const float* __restrict__ g,
                              const float* __restrict__ b, int C, float Minv,
                              float* __restrict__ scsh) {
    int c = blockIdx.x * blockDim.x + threadIdx.x;
    if (c >= C) return;
    float mean = sums[c] * Minv;
    float var  = sums[C + c] * Minv - mean * mean;
    float sc   = g[c] * rsqrtf(var + BN_EPS);
    scsh[c]     = sc;
    scsh[C + c] = b[c] - mean * sc;
}

__global__ void k_bn_relu_tohalf(const float* __restrict__ x, const float* __restrict__ scsh,
                                 int C, long total, _Float16* __restrict__ out) {
    long i = (long)blockIdx.x * blockDim.x + threadIdx.x;
    if (i >= total) return;
    int c = (int)(i % C);
    float v = x[i] * scsh[c] + scsh[C + c];
    out[i] = (_Float16)fmaxf(v, 0.f);
}

__global__ void k_bn_f32(const float* __restrict__ x, const float* __restrict__ scsh,
                         int C, long total, int relu, float* __restrict__ out) {
    long i = (long)blockIdx.x * blockDim.x + threadIdx.x;
    if (i >= total) return;
    int c = (int)(i % C);
    float v = x[i] * scsh[c] + scsh[C + c];
    out[i] = relu ? fmaxf(v, 0.f) : v;
}

// ------------------------- virtual-node pooling -------------------------

__global__ void k_pool(const float* __restrict__ h, const int* __restrict__ batch,
                       float* __restrict__ pooled) {
    int n = blockIdx.x, c = threadIdx.x;
    atomicAdd(&pooled[batch[n] * DD + c], h[(size_t)n * DD + c]);
}

__global__ void k_pooled_vn_tohalf(const float* __restrict__ pooled, const float* __restrict__ vn,
                                   _Float16* __restrict__ g16) {
    int i = blockIdx.x * DD + threadIdx.x;
    g16[i] = (_Float16)(pooled[i] + vn[i]);
}

// ------------------------- launch -------------------------

static inline unsigned cdiv(long a, long b) { return (unsigned)((a + b - 1) / b); }

extern "C" void kernel_launch(void* const* d_in, const int* in_sizes, int n_in,
                              void* d_out, int out_size, void* d_ws, size_t ws_size,
                              hipStream_t stream) {
    const int*   x        = (const int*)  d_in[0];
    const int*   ei       = (const int*)  d_in[1];
    const int*   ea       = (const int*)  d_in[2];
    const int*   batch    = (const int*)  d_in[3];
    const float* atom_emb = (const float*)d_in[4];
    const float* vn_emb   = (const float*)d_in[5];
    const float* bond_emb = (const float*)d_in[6];
    const float* eps      = (const float*)d_in[7];
    const float* cW1  = (const float*)d_in[8];
    const float* cb1  = (const float*)d_in[9];
    const float* cbng = (const float*)d_in[10];
    const float* cbnb = (const float*)d_in[11];
    const float* cW2  = (const float*)d_in[12];
    const float* cb2  = (const float*)d_in[13];
    const float* bng  = (const float*)d_in[14];
    const float* bnb  = (const float*)d_in[15];
    const float* vW1   = (const float*)d_in[16];
    const float* vb1   = (const float*)d_in[17];
    const float* vbn1g = (const float*)d_in[18];
    const float* vbn1b = (const float*)d_in[19];
    const float* vW2   = (const float*)d_in[20];
    const float* vb2   = (const float*)d_in[21];
    const float* vbn2g = (const float*)d_in[22];
    const float* vbn2b = (const float*)d_in[23];

    // ---- workspace carve-out ----
    char* ws = (char*)d_ws;
    size_t off = 0;
    auto carve = [&](size_t bytes) { void* p = ws + off; off += (bytes + 255) & ~(size_t)255; return p; };
    float*    hbuf   = (float*)   carve((size_t)NN * DD * 4);      // node states
    float*    zbuf   = (float*)   carve((size_t)NN * DD * 4);      // agg / GEMM2 out
    float*    y1     = (float*)   carve((size_t)NN * 2 * DD * 4);  // GEMM1 out
    _Float16* a16    = (_Float16*)carve((size_t)NN * 2 * DD * 2);  // fp16 GEMM A operand
    _Float16* w16    = (_Float16*)carve((size_t)512 * 256 * 2);    // fp16 transposed weights
    float*    sums   = (float*)   carve(2 * 512 * 4);
    float*    scsh   = (float*)   carve(2 * 512 * 4);
    float*    vsums  = (float*)   carve(2 * 512 * 4);
    float*    vscsh  = (float*)   carve(2 * 512 * 4);
    float*    vn     = (float*)   carve((size_t)GG * DD * 4);
    float*    pooled = (float*)   carve((size_t)GG * DD * 4);      // also vy2
    _Float16* g16    = (_Float16*)carve((size_t)GG * 512 * 2);
    float*    vy1    = (float*)   carve((size_t)GG * 512 * 4);
    (void)ws_size; (void)in_sizes; (void)n_in; (void)out_size;

    const long ND  = (long)NN * DD;        // 12.8M
    const long N2D = (long)NN * 2 * DD;    // 25.6M

    // GEMM grids: x = ceil((M/16)/2), y = Nc/128 (4 waves x 32 cols per block)
    const unsigned gxN = (NN / 16 + 1) / 2;   // 1563
    const unsigned gxG = (GG / 16 + 1) / 2;   // 8

    // ---- encoder + vn init ----
    k_atom_encode<<<NN, DD, 0, stream>>>(x, atom_emb, hbuf);
    k_vn_init<<<GG, DD, 0, stream>>>(vn, vn_emb);

    for (int l = 0; l < LL; ++l) {
        // h += vn[batch]
        k_add_vn<<<NN, DD, 0, stream>>>(hbuf, vn, batch);

        // edge messages -> agg (zbuf)
        k_zero<<<cdiv(ND, 256), 256, 0, stream>>>(zbuf, ND);
        k_edge_msg<<<EE, DD, 0, stream>>>(ei, ea, bond_emb + (size_t)l * NBF * BV * DD, hbuf, zbuf);

        // z = (1+eps)h + agg -> fp16
        k_make_z_half<<<cdiv(ND, 256), 256, 0, stream>>>(hbuf, zbuf, eps, l, a16);

        // GEMM1: y1[N,512] = z @ cW1 + cb1
        k_weightT_half<<<cdiv(256L * 512, 256), 256, 0, stream>>>(cW1 + (size_t)l * DD * 2 * DD, w16, DD, 2 * DD);
        k_wmma_gemm<<<dim3(gxN, (2 * DD) / 128), 128, 0, stream>>>(a16, w16, cb1 + l * 2 * DD, y1, NN, DD, 2 * DD);

        // BN1 + relu -> fp16
        k_zero<<<cdiv(2L * 512, 256), 256, 0, stream>>>(sums, 2L * 512);
        k_colstats<<<cdiv(NN, 64), 256, 0, stream>>>(y1, NN, 2 * DD, sums);
        k_bn_finalize<<<cdiv(2 * DD, 256), 256, 0, stream>>>(sums, cbng + l * 2 * DD, cbnb + l * 2 * DD, 2 * DD, 1.f / NN, scsh);
        k_bn_relu_tohalf<<<cdiv(N2D, 256), 256, 0, stream>>>(y1, scsh, 2 * DD, N2D, a16);

        // GEMM2: zbuf[N,256] = t1 @ cW2 + cb2
        k_weightT_half<<<cdiv(512L * 256, 256), 256, 0, stream>>>(cW2 + (size_t)l * 2 * DD * DD, w16, 2 * DD, DD);
        k_wmma_gemm<<<dim3(gxN, DD / 128), 128, 0, stream>>>(a16, w16, cb2 + l * DD, zbuf, NN, 2 * DD, DD);

        // BN2 stats (apply deferred until after vn update, which reads hbuf)
        k_zero<<<cdiv(2L * 256, 256), 256, 0, stream>>>(sums, 2L * 256);
        k_colstats<<<cdiv(NN, 64), 256, 0, stream>>>(zbuf, NN, DD, sums);
        k_bn_finalize<<<1, 256, 0, stream>>>(sums, bng + l * DD, bnb + l * DD, DD, 1.f / NN, scsh);

        if (l < LL - 1) {
            // virtual-node update from pre-conv h
            k_zero<<<cdiv((long)GG * DD, 256), 256, 0, stream>>>(pooled, (long)GG * DD);
            k_pool<<<NN, DD, 0, stream>>>(hbuf, batch, pooled);
            k_pooled_vn_tohalf<<<GG, DD, 0, stream>>>(pooled, vn, g16);

            k_weightT_half<<<cdiv(256L * 512, 256), 256, 0, stream>>>(vW1 + (size_t)l * DD * 2 * DD, w16, DD, 2 * DD);
            k_wmma_gemm<<<dim3(gxG, (2 * DD) / 128), 128, 0, stream>>>(g16, w16, vb1 + l * 2 * DD, vy1, GG, DD, 2 * DD);

            k_zero<<<cdiv(2L * 512, 256), 256, 0, stream>>>(vsums, 2L * 512);
            k_colstats<<<cdiv(GG, 64), 256, 0, stream>>>(vy1, GG, 2 * DD, vsums);
            k_bn_finalize<<<cdiv(2 * DD, 256), 256, 0, stream>>>(vsums, vbn1g + l * 2 * DD, vbn1b + l * 2 * DD, 2 * DD, 1.f / GG, vscsh);
            k_bn_relu_tohalf<<<cdiv((long)GG * 2 * DD, 256), 256, 0, stream>>>(vy1, vscsh, 2 * DD, (long)GG * 2 * DD, g16);

            k_weightT_half<<<cdiv(512L * 256, 256), 256, 0, stream>>>(vW2 + (size_t)l * 2 * DD * DD, w16, 2 * DD, DD);
            k_wmma_gemm<<<dim3(gxG, DD / 128), 128, 0, stream>>>(g16, w16, vb2 + l * DD, pooled, GG, 2 * DD, DD);

            k_zero<<<cdiv(2L * 256, 256), 256, 0, stream>>>(vsums, 2L * 256);
            k_colstats<<<cdiv(GG, 64), 256, 0, stream>>>(pooled, GG, DD, vsums);
            k_bn_finalize<<<1, 256, 0, stream>>>(vsums, vbn2g + l * DD, vbn2b + l * DD, DD, 1.f / GG, vscsh);
            k_bn_f32<<<cdiv((long)GG * DD, 256), 256, 0, stream>>>(pooled, vscsh, DD, (long)GG * DD, 1, vn);

            // now overwrite h with relu(bn(z))
            k_bn_f32<<<cdiv(ND, 256), 256, 0, stream>>>(zbuf, scsh, DD, ND, 1, hbuf);
        } else {
            // last layer: no relu, write result
            k_bn_f32<<<cdiv(ND, 256), 256, 0, stream>>>(zbuf, scsh, DD, ND, 0, (float*)d_out);
        }
    }
}
